// MambaModel_87110526698212
// MI455X (gfx1250) — matrix operand order repaired and checked by reading, not compiled, not run on hardware
//
#include <hip/hip_runtime.h>

// ---------------- problem dims (match reference) ----------------
#define B_  2
#define L_  2048
#define DM  512
#define DI  1024
#define NS  16          // d_state
#define RK  32          // dt_rank
#define RKP 64          // dt_rank zero-padded to one K-step
#define NL  4
#define V_  32000
#define KC  4           // d_conv
#define M_  (B_ * L_)   // 4096 rows of activations

// GEMM tiling
#define KSTEP 64
#define LDSP  72        // LDS row pitch in u16: 64 data + 4-DWORD pad = 144 B (16B aligned)
#define TILE_U16 (64 * LDSP)

typedef unsigned short u16;
typedef __attribute__((ext_vector_type(16))) __bf16 v16bf;
typedef __attribute__((ext_vector_type(8)))  __bf16 v8bf;
typedef __attribute__((ext_vector_type(8)))  float  v8f;
typedef __attribute__((ext_vector_type(4)))  unsigned int u32x4;
typedef __attribute__((ext_vector_type(8)))  int i32x8;
typedef __attribute__((ext_vector_type(4)))  int i32x4;

#if defined(__has_builtin)
#  if __has_builtin(__builtin_amdgcn_tensor_load_to_lds) && __has_builtin(__builtin_amdgcn_s_wait_tensorcnt)
#    define HAVE_TDM 1
#  endif
#endif
#ifndef HAVE_TDM
#  define HAVE_TDM 0
#endif

__device__ __forceinline__ u16 f32_to_bf16(float f) {
    unsigned int u = __builtin_bit_cast(unsigned int, f);
    u += 0x7fffu + ((u >> 16) & 1u);     // round-to-nearest-even
    return (u16)(u >> 16);
}

__device__ __forceinline__ float silu(float x) {
    return x / (1.0f + __expf(-x));
}

// ---------------- generic f32 -> bf16 convert ----------------
__global__ void __launch_bounds__(256) cvt_bf16_kernel(const float* __restrict__ s,
                                                       u16* __restrict__ d, long n) {
    long i = (long)blockIdx.x * blockDim.x + threadIdx.x;
    if (i < n) d[i] = f32_to_bf16(s[i]);
}

// ---------------- f32 [rows,RK] -> bf16 [rows,RKP] zero-padded ----------------
__global__ void __launch_bounds__(256) cvt_pad_dtw_kernel(const float* __restrict__ s,
                                                          u16* __restrict__ d, long nrows) {
    long idx = (long)blockIdx.x * blockDim.x + threadIdx.x;   // 0 .. nrows*RKP
    long r = idx / RKP;
    int  j = (int)(idx % RKP);
    if (r < nrows) d[idx] = (j < RK) ? f32_to_bf16(s[r * RK + j]) : (u16)0;
}

// ---------------- embedding gather ----------------
__global__ void __launch_bounds__(256) embed_kernel(const int* __restrict__ ids,
                                                    const float* __restrict__ emb,
                                                    float* __restrict__ res) {
    int row = blockIdx.x;                 // 0..M_-1
    int t   = threadIdx.x;
    long src = (long)ids[row] * DM;
    long dst = (long)row * DM;
    res[dst + 2 * t]     = emb[src + 2 * t];
    res[dst + 2 * t + 1] = emb[src + 2 * t + 1];
}

// ---------------- residual add + LayerNorm -> bf16 ----------------
__global__ void __launch_bounds__(256) add_ln_kernel(float* __restrict__ res,
                                                     const float* __restrict__ add,
                                                     const float* __restrict__ w,
                                                     const float* __restrict__ b,
                                                     u16* __restrict__ out) {
    __shared__ float s1[256], s2[256];
    int  row = blockIdx.x;
    int  t   = threadIdx.x;
    int  j0  = 2 * t, j1 = 2 * t + 1;
    long base = (long)row * DM;
    float x0 = res[base + j0], x1 = res[base + j1];
    if (add) {
        x0 += add[base + j0];
        x1 += add[base + j1];
        res[base + j0] = x0;
        res[base + j1] = x1;
    }
    s1[t] = x0 + x1;
    s2[t] = x0 * x0 + x1 * x1;
    __syncthreads();
    for (int st = 128; st > 0; st >>= 1) {
        if (t < st) { s1[t] += s1[t + st]; s2[t] += s2[t + st]; }
        __syncthreads();
    }
    float mean = s1[0] * (1.0f / DM);
    float var  = s2[0] * (1.0f / DM) - mean * mean;
    float rs   = rsqrtf(var + 1e-5f);
    out[base + j0] = f32_to_bf16((x0 - mean) * rs * w[j0] + b[j0]);
    out[base + j1] = f32_to_bf16((x1 - mean) * rs * w[j1] + b[j1]);
}

// =====================================================================
//  WMMA bf16 GEMM:  C[M,N] = A[M,K] * W[N,K]^T (+bias)
//  128 threads = 4 wave32s, 64x64 tile, K stepped by 64.
//  Double-buffered Tensor Data Mover staging (tensor_load_to_lds,
//  TENSORcnt-tracked, DMA overlapped with WMMA); ds_store fallback.
// =====================================================================

#if HAVE_TDM
// Issue one TDM 2D tile load: 64 rows x 64 bf16, row stride = K elements,
// LDS destination padded by 4 DWORDs every 32 DWORDs (pitch 144 B = LDSP u16).
__device__ __forceinline__ void tdm_load_tile(const u16* gptr, unsigned ldsOff, int K) {
    unsigned long long ga = (unsigned long long)gptr;
    u32x4 g0;
    g0[0] = 1u;                                           // count=1, user mode
    g0[1] = ldsOff;                                       // lds_addr
    g0[2] = (unsigned)ga;                                 // global_addr[31:0]
    g0[3] = (unsigned)((ga >> 32) & 0x01FFFFFFull)        // global_addr[56:32]
          | (2u << 30);                                   // type = 2 (image)
    i32x8 g1;
    g1[0] = (int)((1u << 16)        // data_size = 1 (2 bytes)
          |      (1u << 20)        // pad_enable
          |      (4u << 22)        // pad_interval: every 32 DWORDs (=128 B row)
          |      (3u << 25));      // pad_amount: 4 DWORDs (row pitch -> 144 B)
    g1[1] = (int)(((unsigned)K & 0xFFFFu) << 16);         // tensor_dim0[15:0]
    g1[2] = (int)((((unsigned)K >> 16) & 0xFFFFu)         // tensor_dim0[31:16]
          |       (64u << 16));                           // tensor_dim1[15:0] = 64 rows
    g1[3] = (int)((unsigned)KSTEP << 16);                 // tile_dim0 = 64
    g1[4] = (int)(64u);                                   // tile_dim1 = 64, tile_dim2 = 0
    g1[5] = (int)K;                                       // tensor_dim0_stride[31:0]
    g1[6] = 0;                                            // stride hi, dim1_stride lo
    g1[7] = 0;
    i32x4 z4 = {0, 0, 0, 0};
#if __clang_major__ >= 23
    i32x8 z8 = {};
    __builtin_amdgcn_tensor_load_to_lds(g0, g1, z4, z4, z8, 0);
#else
    __builtin_amdgcn_tensor_load_to_lds(g0, g1, z4, z4, 0);
#endif
}
#endif

// A-fragment (16-bit A 16x32): lane<16 holds K 0..7 & 16..23, lane>=16 holds K 8..15 & 24..31.
__device__ __forceinline__ v16bf lds_fragA(const u16* base, int lane) {
    const u16* p = base + (lane & 15) * LDSP + ((lane < 16) ? 0 : 8);
    v8bf a = *(const v8bf*)(p);
    v8bf c = *(const v8bf*)(p + 16);
    v16bf f;
#pragma unroll
    for (int i = 0; i < 8; ++i) { f[i] = a[i]; f[8 + i] = c[i]; }
    return f;
}
// B-fragment (16-bit B 32x16): lanes 0-15 hold K 0..7 & 16..23, lanes 16-31 hold K 8..15 & 24..31, col N = lane&15.
__device__ __forceinline__ v16bf lds_fragB(const u16* base, int lane) {
    const u16* p = base + (lane & 15) * LDSP + ((lane < 16) ? 0 : 8);
    v8bf a = *(const v8bf*)(p);
    v8bf c = *(const v8bf*)(p + 16);
    v16bf f;
#pragma unroll
    for (int i = 0; i < 8; ++i) { f[i] = a[i]; f[8 + i] = c[i]; }
    return f;
}

template <bool HAS_BIAS>
__global__ void __launch_bounds__(128)
gemm_bf16_wmma(const u16* __restrict__ A, const u16* __restrict__ W,
               const float* __restrict__ bias, float* __restrict__ C,
               int M, int N, int K) {
#if HAVE_TDM
    __shared__ u16 As[2 * TILE_U16];
    __shared__ u16 Ws[2 * TILE_U16];
#else
    __shared__ u16 As[TILE_U16];
    __shared__ u16 Ws[TILE_U16];
#endif
    const int tid  = threadIdx.x;
    const int lane = tid & 31;
    const int wave = tid >> 5;
    const int wm   = wave & 1;
    const int wn   = wave >> 1;
    const int m0   = blockIdx.y * 64;
    const int n0   = blockIdx.x * 64;

    v8f acc[2][2] = {};

#if HAVE_TDM
    // -------- double-buffered TDM pipeline --------
    const unsigned ldsA = (unsigned)(size_t)(void*)As;
    const unsigned ldsW = (unsigned)(size_t)(void*)Ws;
    const u16* gA = A + (long)m0 * K;
    const u16* gW = W + (long)n0 * K;

    if (wave == 0) {                       // prologue: stage K-slab 0 into buffer 0
        tdm_load_tile(gA, ldsA, K);
        tdm_load_tile(gW, ldsW, K);
    }
    int buf = 0;
    for (int k0 = 0; k0 < K; k0 += KSTEP) {
        if (wave == 0) {
            if (k0 + KSTEP < K) {
                // stage next K-slab into the other buffer, then wait for the
                // two oldest (current-buffer) loads only: DMA overlaps WMMA.
                unsigned bo = (unsigned)((buf ^ 1) * TILE_U16 * 2);   // bytes
                tdm_load_tile(gA + k0 + KSTEP, ldsA + bo, K);
                tdm_load_tile(gW + k0 + KSTEP, ldsW + bo, K);
                __builtin_amdgcn_s_wait_tensorcnt(2);
            } else {
                __builtin_amdgcn_s_wait_tensorcnt(0);
            }
        }
        __syncthreads();                   // current buffer visible to all waves

        const u16* Ab = As + buf * TILE_U16;
        const u16* Wb = Ws + buf * TILE_U16;
#pragma unroll
        for (int ks = 0; ks < 2; ++ks) {   // two 32-wide K sub-slabs
            v16bf af[2], wf[2];
#pragma unroll
            for (int i = 0; i < 2; ++i)
                af[i] = lds_fragA(Ab + (wm * 32 + i * 16) * LDSP + ks * 32, lane);
#pragma unroll
            for (int j = 0; j < 2; ++j)
                wf[j] = lds_fragB(Wb + (wn * 32 + j * 16) * LDSP + ks * 32, lane);
#pragma unroll
            for (int i = 0; i < 2; ++i)
#pragma unroll
                for (int j = 0; j < 2; ++j)
                    acc[i][j] = __builtin_amdgcn_wmma_f32_16x16x32_bf16(
                        false, af[i], false, wf[j], (short)0, acc[i][j], false, false);
        }
        __syncthreads();                   // all reads done before buffer is re-staged
        buf ^= 1;
    }
#else
    // -------- cooperative ds_store staging fallback --------
    const int srow = tid >> 1;   // 0..63 staging row
    const int shf  = tid & 1;    // 32-u16 half of the 64-wide K slab
    for (int k0 = 0; k0 < K; k0 += KSTEP) {
        {
            const v8bf* ga = (const v8bf*)(A + (long)(m0 + srow) * K + k0 + shf * 32);
            v8bf* la = (v8bf*)(As + srow * LDSP + shf * 32);
            la[0] = ga[0]; la[1] = ga[1]; la[2] = ga[2]; la[3] = ga[3];
            const v8bf* gw = (const v8bf*)(W + (long)(n0 + srow) * K + k0 + shf * 32);
            v8bf* lw = (v8bf*)(Ws + srow * LDSP + shf * 32);
            lw[0] = gw[0]; lw[1] = gw[1]; lw[2] = gw[2]; lw[3] = gw[3];
        }
        __syncthreads();
#pragma unroll
        for (int ks = 0; ks < 2; ++ks) {
            v16bf af[2], wf[2];
#pragma unroll
            for (int i = 0; i < 2; ++i)
                af[i] = lds_fragA(As + (wm * 32 + i * 16) * LDSP + ks * 32, lane);
#pragma unroll
            for (int j = 0; j < 2; ++j)
                wf[j] = lds_fragB(Ws + (wn * 32 + j * 16) * LDSP + ks * 32, lane);
#pragma unroll
            for (int i = 0; i < 2; ++i)
#pragma unroll
                for (int j = 0; j < 2; ++j)
                    acc[i][j] = __builtin_amdgcn_wmma_f32_16x16x32_bf16(
                        false, af[i], false, wf[j], (short)0, acc[i][j], false, false);
        }
        __syncthreads();
    }
#endif

    // epilogue: C/D layout -- lane = n + 16*(m>=8), vgpr v holds row (m&7)
    const int chalf = lane >> 4;
    const int cn    = lane & 15;
#pragma unroll
    for (int j = 0; j < 2; ++j) {
        int   colg = n0 + wn * 32 + j * 16 + cn;
        float bv   = HAS_BIAS ? bias[colg] : 0.0f;
#pragma unroll
        for (int i = 0; i < 2; ++i) {
            long base = (long)(m0 + wm * 32 + i * 16 + chalf * 8) * N + colg;
#pragma unroll
            for (int v = 0; v < 8; ++v)
                C[base + (long)v * N] = acc[i][j][v] + bv;
        }
    }
}

// ---------------- depthwise causal conv1d (K=4) + SiLU ----------------
__global__ void __launch_bounds__(256) conv_silu_kernel(const float* __restrict__ xz,
                                                        const float* __restrict__ cw,
                                                        const float* __restrict__ cb,
                                                        float* __restrict__ uf,
                                                        u16* __restrict__ uh) {
    long idx = (long)blockIdx.x * blockDim.x + threadIdx.x;   // 0 .. M_*DI
    int  row = (int)(idx / DI);
    int  d   = (int)(idx % DI);
    int  l   = row % L_;
    float acc = cb[d];
#pragma unroll
    for (int k = 0; k < KC; ++k) {
        int lk = l + k - (KC - 1);
        if (lk >= 0)
            acc += xz[(long)(row + (lk - l)) * (2 * DI) + d] * cw[d * KC + k];
    }
    float s = silu(acc);
    uf[idx] = s;
    uh[idx] = f32_to_bf16(s);
}

// ---------------- slice xdbl[:, :RK] -> bf16, zero-padded to RKP ----------------
__global__ void __launch_bounds__(256) dtslice_kernel(const float* __restrict__ xdbl,
                                                      u16* __restrict__ dtr) {
    long idx = (long)blockIdx.x * blockDim.x + threadIdx.x;   // 0 .. M_*RKP
    int  row = (int)(idx / RKP);
    int  j   = (int)(idx % RKP);
    dtr[idx] = (j < RK) ? f32_to_bf16(xdbl[(long)row * (RK + 2 * NS) + j]) : (u16)0;
}

// ---------------- softplus(dt_lin + bias) in place ----------------
__global__ void __launch_bounds__(256) softplus_kernel(float* __restrict__ dtlin,
                                                       const float* __restrict__ db) {
    long idx = (long)blockIdx.x * blockDim.x + threadIdx.x;   // 0 .. M_*DI
    int  d   = (int)(idx % DI);
    float x  = dtlin[idx] + db[d];
    dtlin[idx] = (x > 20.0f) ? x : log1pf(__expf(x));
}

// ---------------- selective scan: one thread per (b, d) channel ----------------
__global__ void __launch_bounds__(256) scan_kernel(const float* __restrict__ dt,
                                                   const float* __restrict__ u,
                                                   const float* __restrict__ xdbl,
                                                   const float* __restrict__ A_log,
                                                   const float* __restrict__ Dp,
                                                   float* __restrict__ ys) {
    int idx = blockIdx.x * blockDim.x + threadIdx.x;   // 0 .. B_*DI
    int bb  = idx / DI;
    int d   = idx % DI;
    float Ar[NS], h[NS];
#pragma unroll
    for (int n = 0; n < NS; ++n) { Ar[n] = -__expf(A_log[d * NS + n]); h[n] = 0.0f; }
    float Dv = Dp[d];
    for (int l = 0; l < L_; ++l) {
        long  rb  = (long)(bb * L_ + l);
        float dtv = dt[rb * DI + d];
        float uv  = u[rb * DI + d];
        const float* x = xdbl + rb * (RK + 2 * NS);
        float y = 0.0f;
#pragma unroll
        for (int n = 0; n < NS; ++n) {
            float dA = __expf(dtv * Ar[n]);
            h[n] = dA * h[n] + dtv * x[RK + n] * uv;
            y += h[n] * x[RK + NS + n];
        }
        ys[rb * DI + d] = y + uv * Dv;
    }
}

// ---------------- y = ys * silu(z) -> bf16 ----------------
__global__ void __launch_bounds__(256) gate_kernel(const float* __restrict__ ys,
                                                   const float* __restrict__ xz,
                                                   u16* __restrict__ yh) {
    long idx = (long)blockIdx.x * blockDim.x + threadIdx.x;   // 0 .. M_*DI
    int  row = (int)(idx / DI);
    int  d   = (int)(idx % DI);
    float z  = xz[(long)row * (2 * DI) + DI + d];
    yh[idx]  = f32_to_bf16(ys[idx] * silu(z));
}

// =====================================================================
extern "C" void kernel_launch(void* const* d_in, const int* in_sizes, int n_in,
                              void* d_out, int out_size, void* d_ws, size_t ws_size,
                              hipStream_t stream) {
    const int*   ids        = (const int*)d_in[0];
    const float* emb        = (const float*)d_in[1];
    const float* in_proj_w  = (const float*)d_in[2];
    const float* conv_w     = (const float*)d_in[3];
    const float* conv_b     = (const float*)d_in[4];
    const float* x_proj_w   = (const float*)d_in[5];
    const float* dt_proj_w  = (const float*)d_in[6];
    const float* dt_proj_b  = (const float*)d_in[7];
    const float* A_log      = (const float*)d_in[8];
    const float* Dp         = (const float*)d_in[9];
    const float* out_proj_w = (const float*)d_in[10];
    const float* norm_w     = (const float*)d_in[11];
    const float* norm_b     = (const float*)d_in[12];
    const float* normf_w    = (const float*)d_in[13];
    const float* normf_b    = (const float*)d_in[14];
    const float* dec_w      = (const float*)d_in[15];
    const float* dec_b      = (const float*)d_in[16];
    float*       logits     = (float*)d_out;

    // ---- workspace bump allocator ----
    char*  ws  = (char*)d_ws;
    size_t off = 0;
    auto alloc = [&](size_t bytes) -> char* {
        char* p = ws + off;
        off += (bytes + 255) & ~(size_t)255;
        return p;
    };
    float* res    = (float*)alloc((size_t)M_ * DM * 4);
    float* hcur   = (float*)alloc((size_t)M_ * DM * 4);
    u16*   hn     = (u16*)  alloc((size_t)M_ * DM * 2);
    float* xz     = (float*)alloc((size_t)M_ * 2 * DI * 4);
    float* uf     = (float*)alloc((size_t)M_ * DI * 4);
    u16*   uh     = (u16*)  alloc((size_t)M_ * DI * 2);
    float* xdbl   = (float*)alloc((size_t)M_ * (RK + 2 * NS) * 4);
    u16*   dtr    = (u16*)  alloc((size_t)M_ * RKP * 2);
    float* dtlin  = (float*)alloc((size_t)M_ * DI * 4);
    float* ysb    = (float*)alloc((size_t)M_ * DI * 4);
    u16*   yh     = (u16*)  alloc((size_t)M_ * DI * 2);
    u16*   w_in   = (u16*)  alloc((size_t)NL * 2 * DI * DM * 2);
    u16*   w_x    = (u16*)  alloc((size_t)NL * (RK + 2 * NS) * DI * 2);
    u16*   w_dt   = (u16*)  alloc((size_t)NL * DI * RKP * 2);
    u16*   w_out  = (u16*)  alloc((size_t)NL * DM * DI * 2);
    u16*   w_dec  = (u16*)  alloc((size_t)V_ * DM * 2);

    auto cvt = [&](const float* s, u16* d, long n) {
        cvt_bf16_kernel<<<(unsigned)((n + 255) / 256), 256, 0, stream>>>(s, d, n);
    };
    // convert all GEMM weights to bf16 (recomputed every call: deterministic)
    cvt(in_proj_w,  w_in,  (long)NL * 2 * DI * DM);
    cvt(x_proj_w,   w_x,   (long)NL * (RK + 2 * NS) * DI);
    cvt(out_proj_w, w_out, (long)NL * DM * DI);
    cvt(dec_w,      w_dec, (long)V_ * DM);
    // dt_proj_w: [NL*DI, RK] -> [NL*DI, RKP] zero-padded bf16
    cvt_pad_dtw_kernel<<<(unsigned)(((long)NL * DI * RKP + 255) / 256), 256, 0, stream>>>(
        dt_proj_w, w_dt, (long)NL * DI);

    auto gemm = [&](const u16* A, const u16* W, const float* bias,
                    float* C, int M, int N, int K) {
        dim3 g(N / 64, M / 64);
        if (bias)
            gemm_bf16_wmma<true><<<g, 128, 0, stream>>>(A, W, bias, C, M, N, K);
        else
            gemm_bf16_wmma<false><<<g, 128, 0, stream>>>(A, W, nullptr, C, M, N, K);
    };

    // embedding -> residual stream
    embed_kernel<<<M_, 256, 0, stream>>>(ids, emb, res);

    const int EW = (M_ * DI) / 256;   // elementwise grid over [M_, DI]

    for (int i = 0; i < NL; ++i) {
        // residual add (layers > 0) + pre-norm
        add_ln_kernel<<<M_, 256, 0, stream>>>(res, (i == 0) ? nullptr : hcur,
                                              norm_w + (long)i * DM, norm_b + (long)i * DM, hn);
        // in_proj: [M_,DM] x [2DI,DM]^T -> xz f32
        gemm(hn, w_in + (long)i * 2 * DI * DM, nullptr, xz, M_, 2 * DI, DM);
        // causal depthwise conv + SiLU on first half of xz
        conv_silu_kernel<<<EW, 256, 0, stream>>>(xz, conv_w + (long)i * DI * KC,
                                                 conv_b + (long)i * DI, uf, uh);
        // x_proj: [M_,DI] x [64,DI]^T -> xdbl (dt_r | B | C)
        gemm(uh, w_x + (long)i * (RK + 2 * NS) * DI, nullptr, xdbl, M_, RK + 2 * NS, DI);
        // dt = softplus(dt_r @ dt_proj_w^T + dt_proj_b), K zero-padded to 64
        dtslice_kernel<<<(M_ * RKP) / 256, 256, 0, stream>>>(xdbl, dtr);
        gemm(dtr, w_dt + (long)i * DI * RKP, nullptr, dtlin, M_, DI, RKP);
        softplus_kernel<<<EW, 256, 0, stream>>>(dtlin, dt_proj_b + (long)i * DI);
        // selective scan
        scan_kernel<<<(B_ * DI) / 256, 256, 0, stream>>>(dtlin, uf, xdbl,
                                                         A_log + (long)i * DI * NS,
                                                         Dp + (long)i * DI, ysb);
        // gate by silu(z), emit bf16 for out_proj
        gate_kernel<<<EW, 256, 0, stream>>>(ysb, xz, yh);
        // out_proj: [M_,DI] x [DM,DI]^T -> hcur
        gemm(yh, w_out + (long)i * DM * DI, nullptr, hcur, M_, DM, DI);
    }

    // final residual add + LN, then decoder GEMM with bias -> logits
    add_ln_kernel<<<M_, 256, 0, stream>>>(res, hcur, normf_w, normf_b, hn);
    gemm(hn, w_dec, dec_b, logits, M_, V_, DM);
}
